// SparseKNNNodeAttentionLayer_19000935317550
// MI455X (gfx1250) — matrix operand order repaired and checked by reading, not compiled, not run on hardware
//
#include <hip/hip_runtime.h>
#include <hip/hip_bf16.h>
#include <math.h>

// ---------------------------------------------------------------------------
// Types for WMMA fragments (gfx1250, wave32)
// ---------------------------------------------------------------------------
typedef __attribute__((ext_vector_type(16))) __bf16 v16bf;
typedef __attribute__((ext_vector_type(8)))  float  v8f;
typedef __attribute__((ext_vector_type(4)))  unsigned int u32x4;
typedef __attribute__((ext_vector_type(2)))  unsigned int u32x2;

#define HID 128
#define NHEADS 4
#define DHEAD 32
#define MIDD 256
#define LN_EPS 1e-5f
#define INV_SQRT_DH 0.17677669529663687f

__device__ __forceinline__ unsigned short f2bf(float f) {
  unsigned int u = __float_as_uint(f);
  u += 0x7FFFu + ((u >> 16) & 1u);   // round-to-nearest-even
  return (unsigned short)(u >> 16);
}

__device__ __forceinline__ void atomicMaxF(float* addr, float val) {
  unsigned int* ua = (unsigned int*)addr;
  unsigned int old = *ua;
  while (val > __uint_as_float(old)) {
    unsigned int assumed = old;
    old = atomicCAS(ua, assumed, __float_as_uint(val));
    if (old == assumed) break;
  }
}

union AFrag { v16bf v; u32x4 q[2]; };

// A fragment (16-bit A 16x32 layout): lane<16 -> K {ko..ko+7, ko+16..ko+23},
// lane>=16 -> +8.  ko must be 8-aligned relative to a 16B-aligned row.
template <int LDK>
__device__ __forceinline__ AFrag load_afrag(const unsigned short (*Xs)[LDK],
                                            int row, int ko) {
  AFrag a;
  a.q[0] = *reinterpret_cast<const u32x4*>(&Xs[row][ko]);
  a.q[1] = *reinterpret_cast<const u32x4*>(&Xs[row][ko + 16]);
  return a;
}
// B fragment (32x16): lane<16 -> K {kb..kb+7, kb+16..kb+23} of col N=lane; lane>=16 -> +8.
template <int LDK>
__device__ __forceinline__ AFrag load_bfrag(const unsigned short (*Ws)[LDK],
                                            int col, int ks) {
  AFrag b;
  const int kb = ks - (ks & 16) + ((ks & 16) >> 1);
  b.q[0] = *reinterpret_cast<const u32x4*>(&Ws[col][kb]);
  b.q[1] = *reinterpret_cast<const u32x4*>(&Ws[col][kb + 16]);
  return b;
}

// ---------------------------------------------------------------------------
// Init: mx = -inf, den = 0, agg = 0
// ---------------------------------------------------------------------------
__global__ void init_kernel(float* mx, float* den, float* agg, int N) {
  int i = blockIdx.x * blockDim.x + threadIdx.x;
  int nMx = N * NHEADS;
  if (i < nMx) { mx[i] = -3.402823466e38f; den[i] = 0.0f; }
  for (int j = i; j < N * HID; j += gridDim.x * blockDim.x) agg[j] = 0.0f;
}

// ---------------------------------------------------------------------------
// Fused QKV projection: stage h tile (128 rows x K=128) to LDS once as bf16,
// then run three weight matrices over it.  B fragments are batch-loaded into
// distinct registers so the WMMA chain runs back-to-back behind one wait.
// ---------------------------------------------------------------------------
__global__ void __launch_bounds__(256)
qkv_bf16_kernel(const float* __restrict__ X,
                const float* __restrict__ Wq, const float* __restrict__ Wk,
                const float* __restrict__ Wv,
                float* __restrict__ Q, float* __restrict__ Kp,
                float* __restrict__ V, int M) {
  constexpr int K = 128, NOUT = 128, KT = 64;
  constexpr int LDX = K + 8;    // 136 ushorts: 272B row stride, 16B aligned
  constexpr int LDW = KT + 8;   // 72 ushorts: 144B row stride, 16B aligned
  constexpr int NT = NOUT / 16;

  __shared__ __align__(16) unsigned short Xs[128][LDX];
  __shared__ __align__(16) unsigned short Ws[NOUT][LDW];

  const int tid  = threadIdx.x;
  const int wave = tid >> 5;
  const int lane = tid & 31;
  const int half = lane >> 4;
  const int l16  = lane & 15;
  const int r0   = blockIdx.x * 128;
  const int mbase = wave * 16;
  const int arow  = mbase + l16;

  // stage full X tile (128 x 128) once
  #pragma unroll
  for (int i = 0; i < 16; ++i) {
    int idx4 = tid * 4 + i * 1024;           // 128*128 = 16384 elements
    int row = idx4 >> 7;
    int col = idx4 & 127;
    float4 f = make_float4(0.f, 0.f, 0.f, 0.f);
    if (r0 + row < M)
      f = *reinterpret_cast<const float4*>(&X[(size_t)(r0 + row) * K + col]);
    u32x2 pk;
    pk.x = (unsigned int)f2bf(f.x) | ((unsigned int)f2bf(f.y) << 16);
    pk.y = (unsigned int)f2bf(f.z) | ((unsigned int)f2bf(f.w) << 16);
    *reinterpret_cast<u32x2*>(&Xs[row][col]) = pk;
  }

  const float* Wsel[3] = {Wq, Wk, Wv};
  float* Ysel[3] = {Q, Kp, V};

  for (int w = 0; w < 3; ++w) {
    const float* W = Wsel[w];
    float* Y = Ysel[w];
    v8f acc[NT];
    #pragma unroll
    for (int t = 0; t < NT; ++t) acc[t] = (v8f){0.f,0.f,0.f,0.f,0.f,0.f,0.f,0.f};

    for (int k0 = 0; k0 < K; k0 += KT) {
      __syncthreads();   // previous compute done before Ws overwrite
      for (int idx = tid; idx < NOUT * KT; idx += 256) {
        int n = idx / KT;
        int k = idx % KT;
        Ws[n][k] = f2bf(W[(size_t)(k0 + k) * NOUT + n]);
      }
      __syncthreads();

      #pragma unroll
      for (int kk = 0; kk < KT; kk += 32) {
        AFrag a = load_afrag<LDX>(Xs, arow, k0 + kk + (half ? 8 : 0));
        const int ks = kk + half * 16;
        AFrag b[NT];
        #pragma unroll
        for (int t = 0; t < NT; ++t) b[t] = load_bfrag<LDW>(Ws, t * 16 + l16, ks);
        #pragma unroll
        for (int t = 0; t < NT; ++t)
          acc[t] = __builtin_amdgcn_wmma_f32_16x16x32_bf16(
              false, a.v, false, b[t].v, (short)0, acc[t], false, false);
      }
    }

    #pragma unroll
    for (int t = 0; t < NT; ++t) {
      const int col = t * 16 + l16;
      #pragma unroll
      for (int i = 0; i < 8; ++i) {
        int grow = r0 + mbase + half * 8 + i;
        if (grow < M) Y[(size_t)grow * NOUT + col] = acc[t][i];
      }
    }
  }
}

// ---------------------------------------------------------------------------
// Generic WMMA bf16 GEMM: Y[M,NOUT] = act(X[M,K] @ W[K,NOUT] + bias)
// Block: 256 threads (8 waves); block tile 128 x NOUT; K staged in 64-wide
// LDS tiles; B fragments batch-loaded (groups of <=8) into distinct regs.
// ---------------------------------------------------------------------------
template <int K, int NOUT, int ACT>
__global__ void __launch_bounds__(256)
gemm_bf16_kernel(const float* __restrict__ X, const float* __restrict__ W,
                 const float* __restrict__ bias, float* __restrict__ Y, int M) {
  constexpr int KT = 64;
  constexpr int LDK = KT + 8;
  constexpr int NT = NOUT / 16;
  constexpr int TB = (NT < 8) ? NT : 8;   // B-fragment batch size

  __shared__ __align__(16) unsigned short Xs[128][LDK];
  __shared__ __align__(16) unsigned short Ws[NOUT][LDK];

  const int tid  = threadIdx.x;
  const int wave = tid >> 5;
  const int lane = tid & 31;
  const int half = lane >> 4;
  const int l16  = lane & 15;
  const int r0   = blockIdx.x * 128;
  const int mbase = wave * 16;
  const int arow  = mbase + l16;

  v8f acc[NT];
  #pragma unroll
  for (int t = 0; t < NT; ++t) acc[t] = (v8f){0.f,0.f,0.f,0.f,0.f,0.f,0.f,0.f};

  for (int k0 = 0; k0 < K; k0 += KT) {
    #pragma unroll
    for (int i = 0; i < 8; ++i) {
      int idx4 = tid * 4 + i * 1024;          // 128*64 = 8192 elements
      int row = idx4 >> 6;
      int col = idx4 & 63;
      float4 f = make_float4(0.f, 0.f, 0.f, 0.f);
      if (r0 + row < M)
        f = *reinterpret_cast<const float4*>(&X[(size_t)(r0 + row) * K + k0 + col]);
      u32x2 pk;
      pk.x = (unsigned int)f2bf(f.x) | ((unsigned int)f2bf(f.y) << 16);
      pk.y = (unsigned int)f2bf(f.z) | ((unsigned int)f2bf(f.w) << 16);
      *reinterpret_cast<u32x2*>(&Xs[row][col]) = pk;
    }
    for (int idx = tid; idx < NOUT * KT; idx += 256) {
      int n = idx / KT;
      int k = idx % KT;
      Ws[n][k] = f2bf(W[(size_t)(k0 + k) * NOUT + n]);
    }
    __syncthreads();

    #pragma unroll
    for (int kk = 0; kk < KT; kk += 32) {
      AFrag a = load_afrag<LDK>(Xs, arow, kk + (half ? 8 : 0));
      const int ks = kk + half * 16;
      #pragma unroll
      for (int tb = 0; tb < NT; tb += TB) {
        AFrag b[TB];
        #pragma unroll
        for (int j = 0; j < TB; ++j)
          b[j] = load_bfrag<LDK>(Ws, (tb + j) * 16 + l16, ks);
        #pragma unroll
        for (int j = 0; j < TB; ++j)
          acc[tb + j] = __builtin_amdgcn_wmma_f32_16x16x32_bf16(
              false, a.v, false, b[j].v, (short)0, acc[tb + j], false, false);
      }
    }
    __syncthreads();
  }

  #pragma unroll
  for (int t = 0; t < NT; ++t) {
    const int col = t * 16 + l16;
    float bv = (bias != nullptr) ? bias[col] : 0.0f;
    #pragma unroll
    for (int i = 0; i < 8; ++i) {
      int grow = r0 + mbase + half * 8 + i;
      if (grow < M) {
        float v = acc[t][i] + bv;
        if (ACT == 1) v = v * (1.0f / (1.0f + __expf(-v)));   // SiLU
        Y[(size_t)grow * NOUT + col] = v;
      }
    }
  }
}

// ---------------------------------------------------------------------------
// Edge pass 1: one wave per edge.
// logits[e,h] = dot(q_h, k_h + ke_h)/sqrt(DH) + eb_h ; atomic-max into mx.
// e_out[e,:] = LN(silu(ke+ve)) written directly to output.
// ---------------------------------------------------------------------------
__global__ void __launch_bounds__(256)
edge_logits_kernel(const float* __restrict__ EA, const long long* __restrict__ EI,
                   const float* __restrict__ Q, const float* __restrict__ Km,
                   const float* __restrict__ Wke, const float* __restrict__ Wve,
                   const float* __restrict__ Web,
                   const float* __restrict__ ln_g, const float* __restrict__ ln_b,
                   float* __restrict__ logits, float* __restrict__ mx,
                   float* __restrict__ eout, int E) {
  __shared__ float sKe[7][HID];
  __shared__ float sVe[7][HID];
  __shared__ float sEb[7][NHEADS];
  for (int i = threadIdx.x; i < 7 * HID; i += 256) { sKe[i / HID][i % HID] = Wke[i]; sVe[i / HID][i % HID] = Wve[i]; }
  for (int i = threadIdx.x; i < 7 * NHEADS; i += 256) sEb[i / NHEADS][i % NHEADS] = Web[i];
  __syncthreads();

  const int wave = threadIdx.x >> 5;
  const int lane = threadIdx.x & 31;
  const int e = blockIdx.x * 8 + wave;
  if (e >= E) return;

  const long long cur = EI[e];
  const long long nbr = EI[(size_t)E + e];
  float ea[7];
  #pragma unroll
  for (int d = 0; d < 7; ++d) ea[d] = EA[(size_t)e * 7 + d];

  const int c0 = lane * 4;
  float ke[4], ve[4];
  #pragma unroll
  for (int j = 0; j < 4; ++j) {
    float sk = 0.f, sv = 0.f;
    #pragma unroll
    for (int d = 0; d < 7; ++d) { sk += ea[d] * sKe[d][c0 + j]; sv += ea[d] * sVe[d][c0 + j]; }
    ke[j] = sk; ve[j] = sv;
  }

  float4 qv = *reinterpret_cast<const float4*>(&Q[(size_t)cur * HID + c0]);
  float4 kv = *reinterpret_cast<const float4*>(&Km[(size_t)nbr * HID + c0]);
  float part = qv.x * (kv.x + ke[0]) + qv.y * (kv.y + ke[1]) +
               qv.z * (kv.z + ke[2]) + qv.w * (kv.w + ke[3]);
  part += __shfl_xor(part, 1, 8);
  part += __shfl_xor(part, 2, 8);
  part += __shfl_xor(part, 4, 8);
  const int hd = lane >> 3;   // 8 lanes per head (32 cols)
  if ((lane & 7) == 0) {
    float eb = 0.f;
    #pragma unroll
    for (int d = 0; d < 7; ++d) eb += ea[d] * sEb[d][hd];
    float lg = part * INV_SQRT_DH + eb;
    logits[(size_t)e * NHEADS + hd] = lg;
    atomicMaxF(&mx[(size_t)cur * NHEADS + hd], lg);
  }

  // fused e_out = LN(silu(ke+ve))
  float eo[4], s = 0.f, s2 = 0.f;
  #pragma unroll
  for (int j = 0; j < 4; ++j) {
    float x = ke[j] + ve[j];
    eo[j] = x * (1.0f / (1.0f + __expf(-x)));
    s += eo[j]; s2 += eo[j] * eo[j];
  }
  #pragma unroll
  for (int m = 1; m < 32; m <<= 1) { s += __shfl_xor(s, m, 32); s2 += __shfl_xor(s2, m, 32); }
  float mean = s * (1.0f / HID);
  float var  = s2 * (1.0f / HID) - mean * mean;
  float rstd = rsqrtf(var + LN_EPS);
  float4 o;
  o.x = (eo[0] - mean) * rstd * ln_g[c0 + 0] + ln_b[c0 + 0];
  o.y = (eo[1] - mean) * rstd * ln_g[c0 + 1] + ln_b[c0 + 1];
  o.z = (eo[2] - mean) * rstd * ln_g[c0 + 2] + ln_b[c0 + 2];
  o.w = (eo[3] - mean) * rstd * ln_g[c0 + 3] + ln_b[c0 + 3];
  *reinterpret_cast<float4*>(&eout[(size_t)e * HID + c0]) = o;
}

// ---------------------------------------------------------------------------
// Edge pass 2: ex = exp(logit - mx[cur]); logits <- ex ; den += ex
// ---------------------------------------------------------------------------
__global__ void edge_exp_kernel(const long long* __restrict__ EI,
                                float* __restrict__ logits,
                                const float* __restrict__ mx,
                                float* __restrict__ den, int E) {
  int idx = blockIdx.x * blockDim.x + threadIdx.x;
  if (idx >= E * NHEADS) return;
  int e = idx >> 2;
  int hd = idx & 3;
  long long cur = EI[e];
  float ex = __expf(logits[idx] - mx[(size_t)cur * NHEADS + hd]);
  logits[idx] = ex;
  atomicAdd(&den[(size_t)cur * NHEADS + hd], ex);
}

// ---------------------------------------------------------------------------
// Edge pass 3: msg = (v[nbr] + ve) * alpha ; agg[cur] += msg (atomic)
// ---------------------------------------------------------------------------
__global__ void __launch_bounds__(256)
edge_scatter_kernel(const float* __restrict__ EA, const long long* __restrict__ EI,
                    const float* __restrict__ V, const float* __restrict__ Wve,
                    const float* __restrict__ ex, const float* __restrict__ den,
                    float* __restrict__ agg, int E) {
  __shared__ float sVe[7][HID];
  for (int i = threadIdx.x; i < 7 * HID; i += 256) sVe[i / HID][i % HID] = Wve[i];
  __syncthreads();

  const int wave = threadIdx.x >> 5;
  const int lane = threadIdx.x & 31;
  const int e = blockIdx.x * 8 + wave;
  if (e >= E) return;

  const long long cur = EI[e];
  const long long nbr = EI[(size_t)E + e];
  float ea[7];
  #pragma unroll
  for (int d = 0; d < 7; ++d) ea[d] = EA[(size_t)e * 7 + d];

  const int c0 = lane * 4;
  const int hd = lane >> 3;
  float alpha = ex[(size_t)e * NHEADS + hd] /
                (den[(size_t)cur * NHEADS + hd] + 1e-16f);

  float4 vv = *reinterpret_cast<const float4*>(&V[(size_t)nbr * HID + c0]);
  float ve[4];
  #pragma unroll
  for (int j = 0; j < 4; ++j) {
    float sv = 0.f;
    #pragma unroll
    for (int d = 0; d < 7; ++d) sv += ea[d] * sVe[d][c0 + j];
    ve[j] = sv;
  }
  atomicAdd(&agg[(size_t)cur * HID + c0 + 0], (vv.x + ve[0]) * alpha);
  atomicAdd(&agg[(size_t)cur * HID + c0 + 1], (vv.y + ve[1]) * alpha);
  atomicAdd(&agg[(size_t)cur * HID + c0 + 2], (vv.z + ve[2]) * alpha);
  atomicAdd(&agg[(size_t)cur * HID + c0 + 3], (vv.w + ve[3]) * alpha);
}

// ---------------------------------------------------------------------------
// Rowwise LN(a + b): one wave per row (H=128, 4 vals/lane)
// ---------------------------------------------------------------------------
__global__ void __launch_bounds__(256)
ln_add_kernel(const float* __restrict__ A, const float* __restrict__ B,
              const float* __restrict__ g, const float* __restrict__ b,
              float* __restrict__ out, int M) {
  const int wave = threadIdx.x >> 5;
  const int lane = threadIdx.x & 31;
  const int row = blockIdx.x * 8 + wave;
  if (row >= M) return;
  const int c0 = lane * 4;
  float4 av = *reinterpret_cast<const float4*>(&A[(size_t)row * HID + c0]);
  float4 bv = *reinterpret_cast<const float4*>(&B[(size_t)row * HID + c0]);
  float x[4] = {av.x + bv.x, av.y + bv.y, av.z + bv.z, av.w + bv.w};
  float s = x[0] + x[1] + x[2] + x[3];
  float s2 = x[0]*x[0] + x[1]*x[1] + x[2]*x[2] + x[3]*x[3];
  #pragma unroll
  for (int m = 1; m < 32; m <<= 1) { s += __shfl_xor(s, m, 32); s2 += __shfl_xor(s2, m, 32); }
  float mean = s * (1.0f / HID);
  float var  = s2 * (1.0f / HID) - mean * mean;
  float rstd = rsqrtf(var + LN_EPS);
  float4 o;
  o.x = (x[0] - mean) * rstd * g[c0 + 0] + b[c0 + 0];
  o.y = (x[1] - mean) * rstd * g[c0 + 1] + b[c0 + 1];
  o.z = (x[2] - mean) * rstd * g[c0 + 2] + b[c0 + 2];
  o.w = (x[3] - mean) * rstd * g[c0 + 3] + b[c0 + 3];
  *reinterpret_cast<float4*>(&out[(size_t)row * HID + c0]) = o;
}

// ---------------------------------------------------------------------------
// Host launcher
// ---------------------------------------------------------------------------
extern "C" void kernel_launch(void* const* d_in, const int* in_sizes, int n_in,
                              void* d_out, int out_size, void* d_ws, size_t ws_size,
                              hipStream_t stream) {
  const float*      h   = (const float*)d_in[0];
  const long long*  EI  = (const long long*)d_in[1];
  const float*      EA  = (const float*)d_in[2];
  const float* Wq   = (const float*)d_in[3];
  const float* Wk   = (const float*)d_in[4];
  const float* Wv   = (const float*)d_in[5];
  const float* Wke  = (const float*)d_in[6];
  const float* Wve  = (const float*)d_in[7];
  const float* Web  = (const float*)d_in[8];
  const float* Wo   = (const float*)d_in[9];
  const float* lnag = (const float*)d_in[10];
  const float* lnab = (const float*)d_in[11];
  const float* lnfg = (const float*)d_in[12];
  const float* lnfb = (const float*)d_in[13];
  const float* lneg = (const float*)d_in[14];
  const float* lneb = (const float*)d_in[15];
  const float* W1   = (const float*)d_in[16];
  const float* b1   = (const float*)d_in[17];
  const float* W2   = (const float*)d_in[18];
  const float* b2   = (const float*)d_in[19];

  const int N = in_sizes[0] / HID;
  const int E = in_sizes[2] / 7;

  float* ws = (float*)d_ws;
  size_t off = 0;
  float* Q      = ws + off; off += (size_t)N * HID;
  float* Kp     = ws + off; off += (size_t)N * HID;
  float* Vp     = ws + off; off += (size_t)N * HID;
  float* logits = ws + off; off += (size_t)E * NHEADS;   // reused as ex
  float* mx     = ws + off; off += (size_t)N * NHEADS;
  float* den    = ws + off; off += (size_t)N * NHEADS;
  float* agg    = ws + off; off += (size_t)N * HID;
  float* attn   = ws + off; off += (size_t)N * HID;      // agg @ Wo
  float* h1     = ws + off; off += (size_t)N * HID;
  float* mid    = ws + off; off += (size_t)N * MIDD;
  float* ffn2   = ws + off; off += (size_t)N * HID;

  float* h2   = (float*)d_out;                    // [N, H]
  float* eout = (float*)d_out + (size_t)N * HID;  // [E, H]

  const int gemmBlocks = (N + 127) / 128;
  const int edgeBlocks = (E + 7) / 8;

  // 0) init accumulators
  init_kernel<<<(N * NHEADS + 255) / 256, 256, 0, stream>>>(mx, den, agg, N);

  // 1) fused node projections Q/K/V = h @ {Wq,Wk,Wv}  (WMMA bf16, X staged once)
  qkv_bf16_kernel<<<gemmBlocks, 256, 0, stream>>>(h, Wq, Wk, Wv, Q, Kp, Vp, N);

  // 2) edge passes (segment softmax + message scatter + fused e_out)
  edge_logits_kernel<<<edgeBlocks, 256, 0, stream>>>(EA, EI, Q, Kp, Wke, Wve, Web,
                                                     lneg, lneb, logits, mx, eout, E);
  edge_exp_kernel<<<(E * NHEADS + 255) / 256, 256, 0, stream>>>(EI, logits, mx, den, E);
  edge_scatter_kernel<<<edgeBlocks, 256, 0, stream>>>(EA, EI, Vp, Wve, logits, den, agg, E);

  // 3) output projection + residual LN
  gemm_bf16_kernel<128, 128, 0><<<gemmBlocks, 256, 0, stream>>>(agg, Wo, nullptr, attn, N);
  ln_add_kernel<<<(N + 7) / 8, 256, 0, stream>>>(h, attn, lnag, lnab, h1, N);

  // 4) FFN (SiLU fused into first GEMM) + residual LN -> h2
  gemm_bf16_kernel<128, 256, 1><<<gemmBlocks, 256, 0, stream>>>(h1, W1, b1, mid, N);
  gemm_bf16_kernel<256, 128, 0><<<gemmBlocks, 256, 0, stream>>>(mid, W2, b2, ffn2, N);
  ln_add_kernel<<<(N + 7) / 8, 256, 0, stream>>>(h1, ffn2, lnfg, lnfb, h2, N);
}
